// QuantLinear_26139170963479
// MI455X (gfx1250) — matrix operand order repaired and checked by reading, not compiled, not run on hardware
//
#include <hip/hip_runtime.h>

// ---------------------------------------------------------------------------
// QuantLinear: Y[M,N] = (X[M,K] @ Wq[N,K]^T) * (1/w_scale) + q_bias*(1/b_scale)
// M=8192, N=16384, K=4096.  2*M*N*K ~= 1.1 TFLOP vs ~40us of HBM traffic
// => compute-bound => f16 WMMA (weights are int-valued in [-127,127], exact
// in f16; accumulate f32; apply scales/bias after accumulation).
//
// Block tile 128x256, 8 wave32 waves in a 2x4 grid, 64x64 per wave:
//   16 x v_wmma_f32_16x16x32_f16 per K=32 stage vs 16 ds_load_b128.
// Double-buffered LDS (60 KB), one barrier per K stage; next tile's global
// loads + f32->f16 converts overlap the current stage's WMMAs.
// ---------------------------------------------------------------------------

typedef __attribute__((ext_vector_type(16))) _Float16 v16h;
typedef __attribute__((ext_vector_type(8)))  _Float16 v8h;
typedef __attribute__((ext_vector_type(4)))  _Float16 v4h;
typedef __attribute__((ext_vector_type(8)))  float    v8f;
typedef __attribute__((ext_vector_type(4)))  float    v4f;

constexpr int MT = 4 * 2048;   // rows of X (B*S)
constexpr int NT = 16384;      // DOUT
constexpr int KT = 4096;       // DIN

constexpr int BM  = 128;       // block tile M
constexpr int BN  = 256;       // block tile N
constexpr int BK  = 32;        // K per LDS stage (one WMMA K-step)
constexpr int PAD = 8;
constexpr int LDT = BK + PAD;  // 40 halves = 80 B row stride (16B aligned)

__global__ __launch_bounds__(256)
void qlinear_wmma_f16(const float* __restrict__ X,
                      const float* __restrict__ Wq,
                      const float* __restrict__ w_scale,
                      const float* __restrict__ Bq,
                      const float* __restrict__ b_scale,
                      float* __restrict__ Y)
{
    __shared__ _Float16 sX[2][BM * LDT];   // 2 * 10.0 KB
    __shared__ _Float16 sW[2][BN * LDT];   // 2 * 20.0 KB  (60 KB total)

    // ---- grid swizzle: 8 M-blocks share an N-band for L2 reuse of W ------
    constexpr int nPidM   = MT / BM;    // 64
    constexpr int nPidN   = NT / BN;    // 64
    constexpr int GROUP_M = 8;
    const int pid          = blockIdx.x;
    const int pidsPerGroup = GROUP_M * nPidN;
    const int group        = pid / pidsPerGroup;
    const int firstM       = group * GROUP_M;
    const int remM         = nPidM - firstM;
    const int gsz          = remM < GROUP_M ? remM : GROUP_M;
    const int pidM         = firstM + (pid % gsz);
    const int pidN         = (pid % pidsPerGroup) / gsz;

    const int m0 = pidM * BM;
    const int n0 = pidN * BN;

    const int tid  = threadIdx.x;
    const int lane = tid & 31;
    const int wave = tid >> 5;          // 8 waves: 2 (M) x 4 (N)
    const int wm   = (wave & 1) * 64;   // wave tile: 64 x 64
    const int wn   = (wave >> 1) * 64;

    // staging map: 8 threads per 32-float row, one b128 each; 32 rows/slab
    const int lr = tid >> 3;            // 0..31
    const int lc = (tid & 7) * 4;       // 0,4,...,28

    // WMMA fragment lane decomposition
    const int mrow = lane & 15;         // row/col within 16x16 fragment
    const int hsel = lane >> 4;         // half-wave select

    v8f acc[4][4];
#pragma unroll
    for (int i = 0; i < 4; ++i)
#pragma unroll
        for (int j = 0; j < 4; ++j)
#pragma unroll
            for (int r = 0; r < 8; ++r) acc[i][j][r] = 0.0f;

    const float* xg = X  + (size_t)(m0 + lr) * KT + lc;   // this thread's rows
    const float* wg = Wq + (size_t)(n0 + lr) * KT + lc;

    // ---- prologue: stage first K-chunk into buffer 0 ---------------------
    {
#pragma unroll
        for (int r = 0; r < BM; r += 32) {
            const v4f v = *(const v4f*)(xg + (size_t)r * KT);
            v4h h;
#pragma unroll
            for (int q = 0; q < 4; ++q) h[q] = (_Float16)v[q];
            *(v4h*)&sX[0][(r + lr) * LDT + lc] = h;
        }
#pragma unroll
        for (int r = 0; r < BN; r += 32) {
            const v4f v = *(const v4f*)(wg + (size_t)r * KT);
            v4h h;
#pragma unroll
            for (int q = 0; q < 4; ++q) h[q] = (_Float16)v[q];
            *(v4h*)&sW[0][(r + lr) * LDT + lc] = h;
        }
    }
    __syncthreads();

    int buf = 0;
    for (int kc = 0; kc < KT; kc += BK) {
        const bool hasNext = (kc + BK) < KT;

        // ---- issue next chunk's global loads early (hide under WMMAs) ---
        v4h xr[BM / 32];
        v4h wr[BN / 32];
        if (hasNext) {
#pragma unroll
            for (int s = 0; s < BM / 32; ++s) {
                const v4f v = *(const v4f*)(xg + (size_t)(s * 32) * KT + (kc + BK));
#pragma unroll
                for (int q = 0; q < 4; ++q) xr[s][q] = (_Float16)v[q];
            }
#pragma unroll
            for (int s = 0; s < BN / 32; ++s) {
                const v4f v = *(const v4f*)(wg + (size_t)(s * 32) * KT + (kc + BK));
#pragma unroll
                for (int q = 0; q < 4; ++q) wr[s][q] = (_Float16)v[q];
            }
        }

        // ---- compute current stage: 16 WMMAs, 16 ds_load_b128 -----------
        {
            v16h a[4];
            v16h b[4];
            // A 16x32 f16: lanes 0-15 K {0..7,16..23}; lanes 16-31 K {8..15,24..31}
#pragma unroll
            for (int i = 0; i < 4; ++i) {
                const _Float16* p = &sX[buf][(wm + i * 16 + mrow) * LDT + hsel * 8];
                v8h lo = *(const v8h*)(p);
                v8h hi = *(const v8h*)(p + 16);
                a[i] = __builtin_shufflevector(lo, hi, 0, 1, 2, 3, 4, 5, 6, 7,
                                               8, 9, 10, 11, 12, 13, 14, 15);
            }
            // B 32x16 f16: lanes 0-15 col N=lane K {0..7,16..23}; lanes 16-31 K {8..15,24..31}
#pragma unroll
            for (int j = 0; j < 4; ++j) {
                const _Float16* p = &sW[buf][(wn + j * 16 + mrow) * LDT + hsel * 8];
                v8h lo = *(const v8h*)(p);
                v8h hi = *(const v8h*)(p + 16);
                b[j] = __builtin_shufflevector(lo, hi, 0, 1, 2, 3, 4, 5, 6, 7,
                                               8, 9, 10, 11, 12, 13, 14, 15);
            }
#pragma unroll
            for (int i = 0; i < 4; ++i)
#pragma unroll
                for (int j = 0; j < 4; ++j)
                    acc[i][j] = __builtin_amdgcn_wmma_f32_16x16x32_f16(
                        /*neg_a=*/false, a[i], /*neg_b=*/false, b[j],
                        /*c_mod=*/(short)0, acc[i][j],
                        /*reuse_a=*/false, /*reuse_b=*/false);
        }

        // ---- write next stage into the other buffer ----------------------
        if (hasNext) {
            const int nb = buf ^ 1;
#pragma unroll
            for (int s = 0; s < BM / 32; ++s)
                *(v4h*)&sX[nb][(s * 32 + lr) * LDT + lc] = xr[s];
#pragma unroll
            for (int s = 0; s < BN / 32; ++s)
                *(v4h*)&sW[nb][(s * 32 + lr) * LDT + lc] = wr[s];
        }
        __syncthreads();
        buf ^= 1;
    }

    // ---- epilogue: dequant scale + bias, non-temporal stores -------------
    const float inv_w = 1.0f / w_scale[0];
    const float inv_b = 1.0f / b_scale[0];

#pragma unroll
    for (int j = 0; j < 4; ++j) {
        const int col  = n0 + wn + j * 16 + mrow;
        const float bv = Bq[col] * inv_b;
#pragma unroll
        for (int i = 0; i < 4; ++i) {
            const int mbase = m0 + wm + i * 16 + hsel * 8;   // VGPR r -> row mbase+r
#pragma unroll
            for (int r = 0; r < 8; ++r) {
                const float out = acc[i][j][r] * inv_w + bv;
                __builtin_nontemporal_store(out, Y + (size_t)(mbase + r) * NT + col);
            }
        }
    }
}

extern "C" void kernel_launch(void* const* d_in, const int* in_sizes, int n_in,
                              void* d_out, int out_size, void* d_ws, size_t ws_size,
                              hipStream_t stream) {
    (void)in_sizes; (void)n_in; (void)out_size; (void)d_ws; (void)ws_size;
    const float* X  = (const float*)d_in[0];   // x         [4,2048,4096] f32
    const float* Wq = (const float*)d_in[1];   // q_weight  [16384,4096]  f32 (int-valued)
    const float* ws = (const float*)d_in[2];   // w_scale   scalar
    const float* Bq = (const float*)d_in[3];   // q_bias    [16384]
    const float* bs = (const float*)d_in[4];   // b_scale   scalar
    float* Y = (float*)d_out;                  // [4,2048,16384] f32

    dim3 grid((MT / BM) * (NT / BN));          // 64 * 64 = 4096 blocks
    dim3 block(256);                           // 8 wave32 waves
    hipLaunchKernelGGL(qlinear_wmma_f16, grid, block, 0, stream,
                       X, Wq, ws, Bq, bs, Y);
}
